// GlobalConvolutionalLayer_32409823216179
// MI455X (gfx1250) — matrix operand order repaired and checked by reading, not compiled, not run on hardware
//
#include <hip/hip_runtime.h>

typedef __attribute__((ext_vector_type(16))) _Float16 v16h;
typedef __attribute__((ext_vector_type(2)))  __fp16   pkh2;  // type returned by cvt_pkrtz
typedef __attribute__((ext_vector_type(8)))  float    v8f;

namespace {
constexpr int   GN     = 4096;   // grid points
constexpr int   BATCH  = 16;     // batch == WMMA M
constexpr int   CM1    = 15;     // channels minus the raw-density channel
constexpr float LOG2E  = 1.4426950408889634f;
}

// One wave computes one 16(batch) x 16(m) tile of out[:, ch, :] by marching the
// contraction dim l in chunks of 32 through v_wmma_f32_16x16x32_f16.
// A = (density*gw) in f16 (staged in LDS), B = Gaussian kernel tile generated
// on the fly with v_exp_f32 and packed with v_cvt_pk_rtz_f16_f32.
__global__ __launch_bounds__(256) void gconv_wmma(
    const float* __restrict__ density, const float* __restrict__ xi,
    const float* __restrict__ grid,    const float* __restrict__ gw,
    float* __restrict__ out)
{
  __shared__ float    sg[GN];           // grid, f32          (16 KB)
  __shared__ _Float16 sw[BATCH * GN];   // W = density*gw f16 (128 KB)

  const int tid = threadIdx.x;
  const int ch  = blockIdx.x >> 5;          // 0..14
  const int m0  = (blockIdx.x & 31) * 128;  // 32 m-groups of 128 columns

  __builtin_prefetch(grid + tid * 4, 0, 0);          // global_prefetch_b8

  // ---- stage grid (b128 copies) ----
  for (int k = tid; k < GN / 4; k += 256)
    ((float4*)sg)[k] = ((const float4*)grid)[k];

  // ---- stage W = density*gw as f16, packed 4-wide per thread ----
  for (int k = tid; k < (BATCH * GN) / 4; k += 256) {
    const int g4 = k & (GN / 4 - 1);
    const float4 d4 = ((const float4*)density)[k];   // row-major (16,4096)
    const float4 w4 = ((const float4*)gw)[g4];
    union { pkh2 p[2]; uint2 u; } pk;
    pk.p[0] = __builtin_amdgcn_cvt_pkrtz(d4.x * w4.x, d4.y * w4.y);
    pk.p[1] = __builtin_amdgcn_cvt_pkrtz(d4.z * w4.z, d4.w * w4.w);
    *(uint2*)&sw[4 * k] = pk.u;                      // ds_store_b64
  }
  __syncthreads();

  // 1/sigmoid(x) = 1 + e^{-x}
  const float xiv  = xi[ch];
  const float xi_t = 1.0f + __builtin_amdgcn_exp2f(-xiv * LOG2E);
  const float nxi  = -xi_t * LOG2E;         // exp(-xi_t*d^2) = exp2(nxi*d^2)
  const float scl  = 0.5f * xi_t;

  const int wave = tid >> 5;                // 8 waves -> 8 m-tiles
  const int lane = tid & 31;
  const int hi   = lane >> 4;               // lane group (K-half selector)
  const int ln   = lane & 15;               // row M (A) / column N (B,C,D)
  const int m    = m0 + wave * 16 + ln;     // this lane's output column
  const float gm = sg[m];

  v8f c = {0.f, 0.f, 0.f, 0.f, 0.f, 0.f, 0.f, 0.f};

#pragma unroll 2
  for (int l = 0; l < GN; l += 32) {
    // ---- A fragment (16x32 f16), row M=ln.
    // lanes 0-15: K = l+0..7 and l+16..23 ; lanes 16-31: +8 on each.
    union { uint4 q[2]; v16h h; } af;
    af.q[0] = *(const uint4*)&sw[ln * GN + l + hi * 8];
    af.q[1] = *(const uint4*)&sw[ln * GN + l + 16 + hi * 8];

    // ---- B fragment (32x16 f16), column N=ln.
    // lanes 0-15: K = l+0..7 and l+16..23 ; lanes 16-31: +8 on each.
    const float4* gp  = (const float4*)&sg[l + hi * 8];
    const float4* gp2 = (const float4*)&sg[l + 16 + hi * 8];
    const float4 q0 = gp[0], q1 = gp[1], q2 = gp2[0], q3 = gp2[1];
    const float gl[16] = {q0.x, q0.y, q0.z, q0.w, q1.x, q1.y, q1.z, q1.w,
                          q2.x, q2.y, q2.z, q2.w, q3.x, q3.y, q3.z, q3.w};
    float e[16];
#pragma unroll
    for (int j = 0; j < 16; ++j) {
      const float d = gl[j] - gm;
      e[j] = __builtin_amdgcn_exp2f(nxi * d * d) * scl;   // v_exp_f32
    }
    union { pkh2 p[8]; v16h h; } bf;
#pragma unroll
    for (int v = 0; v < 8; ++v)
      bf.p[v] = __builtin_amdgcn_cvt_pkrtz(e[2 * v], e[2 * v + 1]);

    // D = A x B + C   (8 args: neg_a, A, neg_b, B, c_mod, C, reuse_a, reuse_b)
    c = __builtin_amdgcn_wmma_f32_16x16x32_f16(false, af.h, false, bf.h,
                                               (short)0, c, false, false);
  }

  // C/D layout: VGPR r holds row r (lanes 0-15) and row r+8 (lanes 16-31).
#pragma unroll
  for (int r = 0; r < 8; ++r) {
    const int b = r + 8 * hi;
    out[(size_t)b * (16 * GN) + (size_t)ch * GN + m] = c[r];
  }
}

// Channel 15 is the raw density passthrough.
__global__ __launch_bounds__(256) void gconv_copy(const float* __restrict__ density,
                                                  float* __restrict__ out)
{
  const int k = blockIdx.x * 256 + threadIdx.x;   // 0..65535
  const int b = k >> 12, g = k & (GN - 1);
  out[(size_t)b * (16 * GN) + (size_t)CM1 * GN + g] = density[k];
}

extern "C" void kernel_launch(void* const* d_in, const int* in_sizes, int n_in,
                              void* d_out, int out_size, void* d_ws, size_t ws_size,
                              hipStream_t stream) {
  const float* density = (const float*)d_in[0];  // (16,1,4096)
  const float* xi      = (const float*)d_in[1];  // (15,)
  const float* grid    = (const float*)d_in[2];  // (4096,)
  const float* gw      = (const float*)d_in[3];  // (4096,)
  float* out = (float*)d_out;                    // (16,16,4096) f32

  // 15 channels x 32 m-groups of (8 waves x 16 cols)
  gconv_wmma<<<CM1 * 32, 256, 0, stream>>>(density, xi, grid, gw, out);
  // passthrough channel
  gconv_copy<<<(BATCH * GN) / 256, 256, 0, stream>>>(density, out);
}
